// WaveFormer_54331336295016
// MI455X (gfx1250) — matrix operand order repaired and checked by reading, not compiled, not run on hardware
//
#include <hip/hip_runtime.h>
#include <hip/hip_bf16.h>

// ---------------- problem constants ----------------
#define B_   4
#define C_   192
#define H_   128
#define W_   128
#define HW_  (H_ * W_)            // 16384
#define NPIX (B_ * HW_)           // 65536
#define C2_  (2 * C_)             // 384

typedef __bf16 bf16;
typedef __attribute__((ext_vector_type(16))) __bf16 v16bf;
typedef __attribute__((ext_vector_type(8)))  __bf16 v8bf;
typedef __attribute__((ext_vector_type(2)))  __bf16 v2bf;
typedef __attribute__((ext_vector_type(8)))  float  v8f;
typedef unsigned u32x4 __attribute__((ext_vector_type(4)));
typedef int      i32x8 __attribute__((ext_vector_type(8)));
typedef int      i32x4 __attribute__((ext_vector_type(4)));

static __device__ __forceinline__ v16bf cat8(v8bf lo, v8bf hi) {
  return __builtin_shufflevector(lo, hi, 0,1,2,3,4,5,6,7,8,9,10,11,12,13,14,15);
}

static __device__ __forceinline__ unsigned pack_bf16(float x, float y) {
  v2bf h = { (__bf16)x, (__bf16)y };
  return __builtin_bit_cast(unsigned, h);   // v_cvt_pk_bf16_f32
}

static __device__ __forceinline__ v8f wmma_bf16(v16bf a, v16bf b, v8f c) {
  return __builtin_amdgcn_wmma_f32_16x16x32_bf16(false, a, false, b,
                                                 (short)0, c, false, false);
}

// ---------------- Tensor Data Mover (gfx1250) ----------------
#if defined(__HIP_DEVICE_COMPILE__) && __has_builtin(__builtin_amdgcn_tensor_load_to_lds)
#define HAVE_TDM 1
#else
#define HAVE_TDM 0
#endif

#if HAVE_TDM
// 2D bf16 tile load Global->LDS via TDM descriptor (D#).
// Tile = tileRows x 32 elements, element = 2 bytes, LDS row pitch padded by
// pad_amount=4 DWORDs every pad_interval=16 DWORDs -> 80B pitch == KPAD halves.
static __device__ __forceinline__ void tdm_load_2d(const void* gptr, unsigned ldsByte,
                                                   int tensorRows, int tileRows, int Kdim) {
  unsigned long long ga = (unsigned long long)gptr;
  u32x4 g0;
  g0.x = 1u;                                                // count=1 (valid user D#)
  g0.y = ldsByte;                                           // lds_addr
  g0.z = (unsigned)ga;                                      // global_addr[31:0]
  g0.w = (unsigned)((ga >> 32) & 0x01FFFFFFu) | (2u << 30); // addr[56:32] | type=2
  i32x8 g1;
  g1[0] = (int)((1u << 16)      // data_size = 2B
              | (1u << 20)      // pad_enable
              | (3u << 22)      // pad_interval: 16 DWORDs (one 64B tile row)
              | (3u << 25));    // pad_amount: 4 DWORDs (16B) -> 80B LDS pitch
  g1[1] = (int)(((unsigned)Kdim & 0xFFFFu) << 16);                       // tensor_dim0 lo
  g1[2] = (int)((((unsigned)Kdim >> 16) & 0xFFFFu)
              | (((unsigned)tensorRows & 0xFFFFu) << 16));               // dim0 hi | dim1 lo
  g1[3] = (int)((((unsigned)tensorRows >> 16) & 0xFFFFu) | (32u << 16)); // dim1 hi | tile_dim0=32
  g1[4] = (int)((unsigned)tileRows & 0xFFFFu);                           // tile_dim1 | tile_dim2=0
  g1[5] = Kdim;                                                          // tensor_dim0_stride lo32
  g1[6] = 0;
  g1[7] = 0;
  i32x4 gz = {0, 0, 0, 0};
#if defined(__clang_major__) && (__clang_major__ >= 23)
  i32x8 gz8 = {0, 0, 0, 0, 0, 0, 0, 0};
  __builtin_amdgcn_tensor_load_to_lds(g0, g1, gz, gz, gz8, 0);
#else
  __builtin_amdgcn_tensor_load_to_lds(g0, g1, gz, gz, 0);
#endif
}
#endif

// ---------------- GEMM tile config ----------------
#define BM 128
#define BN 64
#define BK 32
#define KPAD 40          // 32 + 8 halves pad; 80B row pitch (16B aligned)
#define AHALF (BM * KPAD)   // elements per A LDS buffer
#define BHALF (BN * KPAD)   // elements per B LDS buffer

enum { EPI_STORE = 0, EPI_BIAS = 1, EPI_SPLIT = 2, EPI_GELU = 3, EPI_NCHW = 4 };
enum { TDM_NONE = 0, TDM_A = 1, TDM_B = 2 };   // which operand comes from prepared bf16

struct EpiArgs {
  float*       d0;     // primary output
  float*       d1;     // secondary output (z half for EPI_SPLIT)
  const float* bias;   // bias[colG] (unused for EPI_STORE)
  int          ldd;    // leading dim of d0/d1
  int          split;  // column split point for EPI_SPLIT
};

// D = A(MxK) * B(KxN); bf16 multiplies / fp32 accumulate via
// v_wmma_f32_16x16x32_bf16.  K/LDA/LDB are compile-time so staging offsets
// fold into instruction immediates.  LDS double-buffered (one barrier per
// k-step); the prepared bf16 operand (Wbf, [rows][K]) is DMA'd by the Tensor
// Data Mover into the *next* buffer while WMMAs consume the current one; the
// fp32 operand's global loads are issued before compute and converted/stored
// after, hiding both latencies.  8 waves (4x2), each owns a 32x32 tile.
template<int EPI, int TDM, int K, int LDA, int LDB>
__global__ __launch_bounds__(256)
void gemm_bf16_kernel(const float* __restrict__ Ag, const float* __restrict__ Bg,
                      const bf16* __restrict__ Wbf, int rowsW,
                      long long strideA, long long strideB, long long strideD,
                      EpiArgs ep)
{
  __shared__ __align__(16) bf16 As[2 * AHALF];   // [buf][m][k]
  __shared__ __align__(16) bf16 Bs[2 * BHALF];   // [buf][n][k] (K-transposed)

  const int tid  = threadIdx.x;
  const int wave = tid >> 5;
  const int lane = tid & 31;

  const int mBase = blockIdx.y * BM;
  const int nBase = blockIdx.x * BN;
  const long long zb = blockIdx.z;

  const int wm = (wave >> 1) * 32;
  const int wn = (wave & 1) * 32;
  const int r16  = lane & 15;
  const int hsel = lane >> 4;

  // staging geometry
  const int ar0 = tid >> 4;            // A row slab base, 0..15 (+16j covers 128 rows)
  const int akp = tid & 15;            // A k-pair index, 0..15
  const int bnn = tid & 63;            // B column, 0..63
  const int bkp = tid >> 6;            // B k-pair base, 0..3 (+4j covers 16 pairs)

  const unsigned aLds = (unsigned)(ar0 * KPAD + 2 * akp);
  const unsigned bLds = (unsigned)(bnn * KPAD + 2 * bkp);

  const float* aP = nullptr;
  const float* bP = nullptr;
  if constexpr (TDM != TDM_A)
    aP = Ag + zb * strideA + (unsigned)(mBase + ar0) * (unsigned)LDA + 2u * (unsigned)akp;
  if constexpr (TDM != TDM_B)
    bP = Bg + zb * strideB + (unsigned)(2 * bkp) * (unsigned)LDB + (unsigned)(nBase + bnn);

  float2 ta[8];
  float  tb[8];

  // phase 1: issue global loads (clause-able, no waits between them)
  auto loadA = [&]() {
    #pragma unroll
    for (int j = 0; j < 8; ++j) ta[j] = *(const float2*)(aP + (size_t)(j * 16 * LDA));
    aP += BK;
    __builtin_prefetch(aP, 0, 1);
  };
  auto loadB = [&]() {
    #pragma unroll
    for (int j = 0; j < 4; ++j) {
      tb[2 * j]     = bP[(size_t)(j * 8 * LDB)];
      tb[2 * j + 1] = bP[(size_t)(j * 8 * LDB + LDB)];
    }
    bP += (size_t)(BK * LDB);
    __builtin_prefetch(bP, 0, 1);
  };
  // phase 2: convert + packed store into LDS buffer `buf` (element offset)
  auto storeA = [&](unsigned buf) {
    #pragma unroll
    for (int j = 0; j < 8; ++j)
      *(unsigned*)&As[buf + aLds + j * (16 * KPAD)] = pack_bf16(ta[j].x, ta[j].y);
  };
  auto storeB = [&](unsigned buf) {
    #pragma unroll
    for (int j = 0; j < 4; ++j)
      *(unsigned*)&Bs[buf + bLds + j * 8] = pack_bf16(tb[2 * j], tb[2 * j + 1]);
  };
  // prepared bf16 panel: TDM DMA into buffer `buf` (wave 0 issues)
  auto tdmIssue = [&](int kk, unsigned buf) {
#if HAVE_TDM
    if constexpr (TDM == TDM_A) {
      if (wave == 0)
        tdm_load_2d(Wbf + (unsigned)mBase * (unsigned)K + (unsigned)kk,
                    (unsigned)(unsigned long long)(const void*)&As[0] + buf * 2u,
                    rowsW, BM, K);
    } else if constexpr (TDM == TDM_B) {
      if (wave == 0)
        tdm_load_2d(Wbf + (unsigned)nBase * (unsigned)K + (unsigned)kk,
                    (unsigned)(unsigned long long)(const void*)&Bs[0] + buf * 2u,
                    rowsW, BN, K);
    }
#else
    // fallback: manual bf16 panel copy, identical LDS layout
    if constexpr (TDM == TDM_A) {
      const bf16* s = Wbf + (unsigned)(mBase + ar0) * (unsigned)K
                          + (unsigned)kk + 2u * (unsigned)akp;
      #pragma unroll
      for (int j = 0; j < 8; ++j)
        *(unsigned*)&As[buf + aLds + j * (16 * KPAD)] =
            *(const unsigned*)(s + (size_t)(j * 16 * K));
    } else if constexpr (TDM == TDM_B) {
      const bf16* s = Wbf + (unsigned)(nBase + bnn) * (unsigned)K
                          + (unsigned)kk + 2u * (unsigned)bkp;
      #pragma unroll
      for (int j = 0; j < 4; ++j)
        *(unsigned*)&Bs[buf + bLds + j * 8] = *(const unsigned*)(s + 8u * j);
    }
#endif
  };
  auto tdmWait = [&]() {
#if HAVE_TDM
    if constexpr (TDM != TDM_NONE) {
      if (wave == 0) __builtin_amdgcn_s_wait_tensorcnt(0);
    }
#endif
  };

  v8f acc[4] = { v8f{}, v8f{}, v8f{}, v8f{} };
  constexpr int NSTEPS = K / BK;

  // ---- prologue: stage step 0 into buffer 0 ----
  tdmIssue(0, 0);
  if constexpr (TDM != TDM_A) { loadA(); storeA(0); }
  if constexpr (TDM != TDM_B) { loadB(); storeB(0); }
  tdmWait();
  __syncthreads();

  // ---- pipelined main loop: one barrier per k-step ----
  #pragma unroll
  for (int i = 0; i < NSTEPS; ++i) {
    const unsigned aCur = (unsigned)(i & 1) * AHALF;
    const unsigned bCur = (unsigned)(i & 1) * BHALF;
    const unsigned aNxt = aCur ^ AHALF;
    const unsigned bNxt = bCur ^ BHALF;
    const bool haveNext = (i + 1 < NSTEPS);

    if (haveNext) {
      tdmIssue((i + 1) * BK, (TDM == TDM_A) ? aNxt : bNxt);  // DMA overlaps compute
      if constexpr (TDM != TDM_A) loadA();                   // loads in flight over WMMAs
      if constexpr (TDM != TDM_B) loadB();
    }

    // fragments per ISA 16-bit A/B VGPR layouts (two b128 LDS loads each)
    v16bf af0 = cat8(*(const v8bf*)&As[aCur + (wm + r16) * KPAD + hsel * 8],
                     *(const v8bf*)&As[aCur + (wm + r16) * KPAD + hsel * 8 + 16]);
    v16bf af1 = cat8(*(const v8bf*)&As[aCur + (wm + 16 + r16) * KPAD + hsel * 8],
                     *(const v8bf*)&As[aCur + (wm + 16 + r16) * KPAD + hsel * 8 + 16]);
    v16bf bf0 = cat8(*(const v8bf*)&Bs[bCur + (wn + r16) * KPAD + hsel * 8],
                     *(const v8bf*)&Bs[bCur + (wn + r16) * KPAD + hsel * 8 + 16]);
    v16bf bf1 = cat8(*(const v8bf*)&Bs[bCur + (wn + 16 + r16) * KPAD + hsel * 8],
                     *(const v8bf*)&Bs[bCur + (wn + 16 + r16) * KPAD + hsel * 8 + 16]);

    acc[0] = wmma_bf16(af0, bf0, acc[0]);
    acc[1] = wmma_bf16(af0, bf1, acc[1]);
    acc[2] = wmma_bf16(af1, bf0, acc[2]);
    acc[3] = wmma_bf16(af1, bf1, acc[3]);

    if (haveNext) {
      if constexpr (TDM != TDM_A) storeA(aNxt);
      if constexpr (TDM != TDM_B) storeB(bNxt);
      tdmWait();
    }
    __syncthreads();
  }

  // ---- epilogue (32-bit index math; all flat indices < 2^31) ----
  const int rOff = hsel * 8;
  float biasv[2] = { 0.f, 0.f };
  if (EPI != EPI_STORE) {
    biasv[0] = ep.bias[nBase + wn + r16];
    biasv[1] = ep.bias[nBase + wn + 16 + r16];
  }
  float* d0 = ep.d0 + zb * strideD;

  #pragma unroll
  for (int mi = 0; mi < 2; ++mi) {
    #pragma unroll
    for (int ni = 0; ni < 2; ++ni) {
      const int colG = nBase + wn + ni * 16 + r16;
      #pragma unroll
      for (int r = 0; r < 8; ++r) {
        int   row = mBase + wm + mi * 16 + rOff + r;
        float raw = acc[mi * 2 + ni][r];
        float v   = raw + biasv[ni];
        if (EPI == EPI_STORE) {
          d0[(unsigned)row * (unsigned)ep.ldd + (unsigned)colG] = raw;
        } else if (EPI == EPI_BIAS) {
          d0[(unsigned)row * (unsigned)ep.ldd + (unsigned)colG] = v;
        } else if (EPI == EPI_GELU) {
          d0[(unsigned)row * (unsigned)ep.ldd + (unsigned)colG] =
              0.5f * v * (1.f + erff(v * 0.70710678f));
        } else if (EPI == EPI_SPLIT) {
          if (colG < ep.split) d0[(unsigned)row * (unsigned)ep.ldd + (unsigned)colG] = v;
          else ep.d1[(unsigned)row * (unsigned)ep.ldd + (unsigned)(colG - ep.split)] = v;
        } else { // EPI_NCHW: row = pixel p, colG = channel c -> out[b][c][h][w]
          int bb = row >> 14;            // / HW_
          int hw = row & (HW_ - 1);
          ep.d0[((unsigned)bb * C_ + (unsigned)colG) * HW_ + (unsigned)hw] = v;
        }
      }
    }
  }
}

// ---------------- DCT matrix init (bf16): Wn[n][x], WnT[x][n] ----------------
__global__ void dct_init_kernel(bf16* __restrict__ wn, bf16* __restrict__ wnT) {
  int i = blockIdx.x * 256 + threadIdx.x;
  if (i >= 128 * 128) return;
  int n = i >> 7, x = i & 127;
  float v = __cosf((float)n * ((float)x + 0.5f) * 0.0245436926f) * 0.125f; // pi/128, sqrt(2/128)
  if (n == 0) v *= 0.70710678f;
  wn [n * 128 + x] = (bf16)v;
  wnT[x * 128 + n] = (bf16)v;
}

// ---------------- weight prep: W[k][n] fp32 -> WT[n][k] bf16 ----------------
__global__ void wtrans_bf16_kernel(const float* __restrict__ Wsrc, bf16* __restrict__ WT,
                                   int K, int N) {
  int i = blockIdx.x * 256 + threadIdx.x;
  if (i >= K * N) return;
  int k = i / N, n = i - k * N;          // read coalesced over n
  WT[(unsigned)n * (unsigned)K + (unsigned)k] = (bf16)Wsrc[i];
}

// ---------------- depthwise 3x3 SAME conv, NCHW in -> NHWC out ----------------
__global__ __launch_bounds__(256)
void dwconv_kernel(const float* __restrict__ x, const float* __restrict__ w,
                   const float* __restrict__ b, float* __restrict__ out) {
  long long i = (long long)blockIdx.x * 256 + threadIdx.x; // over B*C*H*W, w fastest
  if (i >= (long long)B_ * C_ * H_ * W_) return;
  int ww = (int)(i & (W_ - 1));
  long long t = i >> 7;
  int hh = (int)(t & (H_ - 1)); t >>= 7;
  int cc = (int)(t % C_);
  int bb = (int)(t / C_);
  const float* xp = x + ((long long)(bb * C_ + cc)) * HW_;
  const float* wp = w + cc * 9;
  float acc = b[cc];
  #pragma unroll
  for (int dy = -1; dy <= 1; ++dy) {
    int h2 = hh + dy;
    if (h2 < 0 || h2 >= H_) continue;
    #pragma unroll
    for (int dx = -1; dx <= 1; ++dx) {
      int w2 = ww + dx;
      if (w2 < 0 || w2 >= W_) continue;
      acc += xp[h2 * W_ + w2] * wp[(dy + 1) * 3 + (dx + 1)];
    }
  }
  out[(((long long)bb * H_ + hh) * W_ + ww) * C_ + cc] = acc;
}

// ---------------- wave modulation: final = cos_t*xu + sin_t*(1+alpha/2)*xu ----
__global__ __launch_bounds__(256)
void modulate_kernel(const float* __restrict__ xu, const float* __restrict__ t,
                     const float* __restrict__ cptr, const float* __restrict__ aptr,
                     float* __restrict__ outF) {
  long long i = (long long)blockIdx.x * 256 + threadIdx.x;
  if (i >= (long long)NPIX * C_) return;
  float c_safe = fabsf(cptr[0]) + 1e-4f;
  float ct = fminf(fmaxf(c_safe * t[i], -20.f), 20.f);
  float cost = __cosf(ct);
  float sint = __sinf(ct) / c_safe;
  float alpha = fmaxf(aptr[0], 0.f);
  float u = xu[i];
  outF[i] = cost * u + sint * (u + 0.5f * alpha * u);
}

// ---------------- LayerNorm over C + silu(z) gate; one wave per pixel ----------
__global__ __launch_bounds__(256)
void ln_gate_kernel(const float* __restrict__ y, const float* __restrict__ z,
                    const float* __restrict__ gamma, const float* __restrict__ beta,
                    float* __restrict__ out) {
  int pix  = blockIdx.x * 8 + (threadIdx.x >> 5);
  int lane = threadIdx.x & 31;
  const float* yp = y + (long long)pix * C_;
  float v[6];
  float s = 0.f, s2 = 0.f;
  #pragma unroll
  for (int j = 0; j < 6; ++j) {
    float t = yp[lane + 32 * j];
    v[j] = t; s += t; s2 += t * t;
  }
  #pragma unroll
  for (int m = 16; m > 0; m >>= 1) {
    s  += __shfl_xor(s,  m, 32);
    s2 += __shfl_xor(s2, m, 32);
  }
  float mu  = s * (1.f / C_);
  float var = s2 * (1.f / C_) - mu * mu;
  float inv = rsqrtf(var + 1e-5f);
  const float* zp = z + (long long)pix * C_;
  float*       op = out + (long long)pix * C_;
  #pragma unroll
  for (int j = 0; j < 6; ++j) {
    int   c  = lane + 32 * j;
    float yn = (v[j] - mu) * inv * gamma[c] + beta[c];
    float zz = zp[c];
    float si = zz / (1.f + __expf(-zz));
    op[c] = yn * si;
  }
}

// ---------------- host-side orchestration ----------------
extern "C" void kernel_launch(void* const* d_in, const int* in_sizes, int n_in,
                              void* d_out, int out_size, void* d_ws, size_t ws_size,
                              hipStream_t stream) {
  const float* x          = (const float*)d_in[0];
  const float* freq_embed = (const float*)d_in[1];
  const float* dw_w       = (const float*)d_in[2];
  const float* dw_b       = (const float*)d_in[3];
  const float* lin_w      = (const float*)d_in[4];
  const float* lin_b      = (const float*)d_in[5];
  const float* tok_w      = (const float*)d_in[6];
  const float* tok_b      = (const float*)d_in[7];
  const float* norm_g     = (const float*)d_in[8];
  const float* norm_b     = (const float*)d_in[9];
  const float* out_w      = (const float*)d_in[10];
  const float* out_b      = (const float*)d_in[11];
  const float* c_s        = (const float*)d_in[12];
  const float* alpha_s    = (const float*)d_in[13];
  float* out = (float*)d_out;

  // workspace layout
  char* ws = (char*)d_ws;
  size_t off = 0;
  auto takeB = [&](size_t bytes) {
    void* p = ws + off;
    off = (off + bytes + 255) & ~(size_t)255;
    return p;
  };
  bf16*  wn_bf   = (bf16*)takeB(128 * 128 * sizeof(bf16));
  bf16*  wnT_bf  = (bf16*)takeB(128 * 128 * sizeof(bf16));
  bf16*  lin_wT  = (bf16*)takeB((size_t)C2_ * C_ * sizeof(bf16));  // [384][192]
  bf16*  tok_wT  = (bf16*)takeB((size_t)C_ * C_ * sizeof(bf16));   // [192][192]
  bf16*  out_wT  = (bf16*)takeB((size_t)C_ * C_ * sizeof(bf16));   // [192][192]
  float* conv    = (float*)takeB((size_t)NPIX * C_ * sizeof(float)); // conv out / gated acts
  float* bx      = (float*)takeB((size_t)NPIX * C_ * sizeof(float)); // x half of linear
  float* bz      = (float*)takeB((size_t)NPIX * C_ * sizeof(float)); // z half of linear
  float* bt      = (float*)takeB((size_t)NPIX * C_ * sizeof(float)); // gelu(freq @ tok_w)
  float* f1      = (float*)takeB((size_t)NPIX * C_ * sizeof(float)); // ping
  float* f2      = (float*)takeB((size_t)NPIX * C_ * sizeof(float)); // pong

  const long long SHWC = (long long)HW_ * C_;   // per-image stride (stage-H batches)
  const long long SWC  = (long long)W_ * C_;    // per-(b,n) stride (stage-W batches)
  const int WC = W_ * C_;                        // 24576

  // 1) prepared bf16 matrices
  dct_init_kernel<<<64, 256, 0, stream>>>(wn_bf, wnT_bf);
  wtrans_bf16_kernel<<<(C_ * C2_ + 255) / 256, 256, 0, stream>>>(lin_w, lin_wT, C_, C2_);
  wtrans_bf16_kernel<<<(C_ * C_  + 255) / 256, 256, 0, stream>>>(tok_w, tok_wT, C_, C_);
  wtrans_bf16_kernel<<<(C_ * C_  + 255) / 256, 256, 0, stream>>>(out_w, out_wT, C_, C_);
  // 2) depthwise conv -> NHWC
  long long ne = (long long)NPIX * C_;
  dwconv_kernel<<<(int)((ne + 255) / 256), 256, 0, stream>>>(x, dw_w, dw_b, conv);
  // 3) xz = conv @ lin_w + lin_b, split into x|z       (B panel via TDM)
  {
    EpiArgs ep{bx, bz, lin_b, C_, C_};
    gemm_bf16_kernel<EPI_SPLIT, TDM_B, C_, C_, 0>
        <<<dim3(C2_ / BN, NPIX / BM, 1), 256, 0, stream>>>(
        conv, nullptr, lin_wT, C2_, 0, 0, 0, ep);
  }
  // 4) t = gelu(freq_embed @ tok_w + tok_b)            (B panel via TDM)
  {
    EpiArgs ep{bt, nullptr, tok_b, C_, 0};
    gemm_bf16_kernel<EPI_GELU, TDM_B, C_, C_, 0>
        <<<dim3(C_ / BN, NPIX / BM, 1), 256, 0, stream>>>(
        freq_embed, nullptr, tok_wT, C_, 0, 0, 0, ep);
  }
  // 5) forward DCT stage H: f1[b] = Wn x X_b[128, W*C] (A panel via TDM, batch B)
  {
    EpiArgs ep{f1, nullptr, nullptr, WC, 0};
    gemm_bf16_kernel<EPI_STORE, TDM_A, H_, 0, W_ * C_>
        <<<dim3(WC / BN, H_ / BM, B_), 256, 0, stream>>>(
        nullptr, bx, wn_bf, H_, 0, SHWC, SHWC, ep);
  }
  // 6) forward DCT stage W: f2[b,n] = Wm x f1[b,n][128, C]  (batch B*H)
  {
    EpiArgs ep{f2, nullptr, nullptr, C_, 0};
    gemm_bf16_kernel<EPI_STORE, TDM_A, W_, 0, C_>
        <<<dim3(C_ / BN, W_ / BM, B_ * H_), 256, 0, stream>>>(
        nullptr, f1, wn_bf, W_, 0, SWC, SWC, ep);
  }
  // 7) modulation: f1 = final(f2, t)
  modulate_kernel<<<(int)((ne + 255) / 256), 256, 0, stream>>>(f2, bt, c_s, alpha_s, f1);
  // 8) inverse stage W: f2[b,n] = WmT x f1[b,n]
  {
    EpiArgs ep{f2, nullptr, nullptr, C_, 0};
    gemm_bf16_kernel<EPI_STORE, TDM_A, W_, 0, C_>
        <<<dim3(C_ / BN, W_ / BM, B_ * H_), 256, 0, stream>>>(
        nullptr, f1, wnT_bf, W_, 0, SWC, SWC, ep);
  }
  // 9) inverse stage H: f1[b] = WnT x f2[b][128, W*C]
  {
    EpiArgs ep{f1, nullptr, nullptr, WC, 0};
    gemm_bf16_kernel<EPI_STORE, TDM_A, H_, 0, W_ * C_>
        <<<dim3(WC / BN, H_ / BM, B_), 256, 0, stream>>>(
        nullptr, f2, wnT_bf, H_, 0, SHWC, SHWC, ep);
  }
  // 10) layernorm + silu(z) gate -> conv (reuse)
  ln_gate_kernel<<<NPIX / 8, 256, 0, stream>>>(f1, bz, norm_g, norm_b, conv);
  // 11) out = g @ out_w + out_b, transposed to NCHW    (B panel via TDM)
  {
    EpiArgs ep{out, nullptr, out_b, 0, 0};
    gemm_bf16_kernel<EPI_NCHW, TDM_B, C_, C_, 0>
        <<<dim3(C_ / BN, NPIX / BM, 1), 256, 0, stream>>>(
        conv, nullptr, out_wT, C_, 0, 0, 0, ep);
  }
}
